// CrossLevel_68118181314634
// MI455X (gfx1250) — compile-verified, matrix operand order checked by reading, not run on hardware
//
#include <hip/hip_runtime.h>

// CDNA5 / gfx1250, wave32.
typedef __attribute__((ext_vector_type(16))) _Float16 v16h;
typedef __attribute__((ext_vector_type(8)))  float    v8f;
typedef __attribute__((ext_vector_type(4)))  float    v4f;

// Reference output is identically zero:
//   A = zeros(N_NODES, N_HYPER);  out = A @ Line_output          -> 0
//   out += 0.0 * sigmoid(...).sum()  (sum finite, in (0, 1e6))   -> +0.0 exact
// Optimal kernel = pure 10.24 MB store stream (~0.44 us at 23.3 TB/s).
// The zero tile is synthesized on the matrix pipe (one
// v_wmma_f32_16x16x32_f16 with zero A/B/C per wave, full EXEC, zero
// marginal cost); each lane's 8xf32 accumulator is exactly two b128
// non-temporal store payloads.
__global__ void __launch_bounds__(256)
crosslevel_zero_out_kernel(float* __restrict__ out, int npairs /* float4 pairs */) {
    v16h a;
#pragma unroll
    for (int i = 0; i < 16; ++i) a[i] = (_Float16)0.0f;
    v8f c = {};
    // D = 0 x 0 + 0 -> exact zeros from the CDNA5 matrix unit.
    c = __builtin_amdgcn_wmma_f32_16x16x32_f16(
        /*neg_a=*/false, a, /*neg_b=*/false, a,
        /*c_mod=*/(short)0, c, /*reuse_a=*/false, /*reuse_b=*/false);

    const v4f z0 = { c[0], c[1], c[2], c[3] };
    const v4f z1 = { c[4], c[5], c[6], c[7] };

    v4f* __restrict__ out4 = reinterpret_cast<v4f*>(out);
    const int stride = gridDim.x * blockDim.x;
    for (int t = blockIdx.x * blockDim.x + threadIdx.x; t < npairs; t += stride) {
        // 32 B per lane per iteration: two NT global_store_b128.
        __builtin_nontemporal_store(z0, &out4[2 * t + 0]);
        __builtin_nontemporal_store(z1, &out4[2 * t + 1]);
    }
}

// Tail (out_size % 8 floats) — not hit for 20000*128, kept for generality.
__global__ void crosslevel_zero_tail_kernel(float* __restrict__ out, int start, int n) {
    int i = start + blockIdx.x * blockDim.x + threadIdx.x;
    if (i < n) out[i] = 0.0f;
}

extern "C" void kernel_launch(void* const* d_in, const int* in_sizes, int n_in,
                              void* d_out, int out_size, void* d_ws, size_t ws_size,
                              hipStream_t stream) {
    (void)d_in; (void)in_sizes; (void)n_in; (void)d_ws; (void)ws_size;

    float* out = reinterpret_cast<float*>(d_out);
    const int npairs = out_size >> 3;        // 320,000 8-float chunks for [20000,128]
    const int threads = 256;                 // 8 waves per block (wave32)
    int blocks = (npairs + threads - 1) / threads;
    if (blocks < 1) blocks = 1;

    if (npairs > 0) {
        crosslevel_zero_out_kernel<<<blocks, threads, 0, stream>>>(out, npairs);
    }
    const int tail_start = npairs << 3;
    const int tail = out_size - tail_start;
    if (tail > 0) {
        crosslevel_zero_tail_kernel<<<1, 64, 0, stream>>>(out, tail_start, out_size);
    }
}
